// PCD_Align_4647154614606
// MI455X (gfx1250) — matrix operand order repaired and checked by reading, not compiled, not run on hardware
//
#include <hip/hip_runtime.h>
#include <hip/hip_bf16.h>
#include <math.h>

// ---------------- types ----------------
typedef __attribute__((ext_vector_type(16))) __bf16 v16bf;
typedef __attribute__((ext_vector_type(8)))  float  v8f;
typedef __attribute__((ext_vector_type(4)))  unsigned int u32x4;
typedef __attribute__((ext_vector_type(8)))  int  i32x8;
typedef __attribute__((ext_vector_type(4)))  int  i32x4;

union AFrag {
    struct { u32x4 lo, hi; } u;
    v16bf v;
};

__device__ __forceinline__ unsigned short f2bf(float f) {
    unsigned int u = __float_as_uint(f);
    u += 0x7FFFu + ((u >> 16) & 1u);   // round-to-nearest-even-ish
    return (unsigned short)(u >> 16);
}

#if defined(__has_builtin)
#if __has_builtin(__builtin_amdgcn_tensor_load_to_lds) && __has_builtin(__builtin_amdgcn_s_wait_tensorcnt)
#define HAVE_TDM 1
#endif
#endif

// ---------------- weight packing: OIHW fp32 -> [Coutpad][9*Cin] bf16 ----------------
// cin_major==0 : k = tap*Cin + cin   (conv im2col order)
// cin_major==1 : k = cin*9  + tap    (DCN einsum order 'ogck')
__global__ __launch_bounds__(256) void pack_w_kernel(
    const float* __restrict__ w, unsigned short* __restrict__ out,
    int Cout, int Cin, int Coutpad, int cin_major)
{
    const int K = 9 * Cin;
    const size_t total = (size_t)Coutpad * K;
    size_t i = (size_t)blockIdx.x * blockDim.x + threadIdx.x;
    if (i >= total) return;
    int k = (int)(i % K);
    int o = (int)(i / K);
    unsigned short v = 0;
    if (o < Cout) {
        int cin, tap;
        if (cin_major) { cin = k / 9;   tap = k - cin * 9; }
        else           { tap = k / Cin; cin = k - tap * Cin; }
        v = f2bf(w[((size_t)o * Cin + cin) * 9 + tap]);
    }
    out[i] = v;
}

// ---------------- NCHW fp32 (virtual concat) -> NHWC bf16 ----------------
__global__ __launch_bounds__(256) void nchw_cat_to_nhwc_kernel(
    const float* __restrict__ xA, const float* __restrict__ xB, int cinA, int Cin,
    unsigned short* __restrict__ out, int Bn, int H, int W)
{
    const int HW = H * W;
    const int VEC = Cin >> 3;
    const size_t total = (size_t)Bn * HW * VEC;
    size_t i = (size_t)blockIdx.x * blockDim.x + threadIdx.x;
    if (i >= total) return;
    const int cv = (int)(i % VEC);
    const size_t p = i / VEC;
    const int b = (int)(p / HW);
    const int s = (int)(p - (size_t)b * HW);
    const int c0 = cv * 8;
    const int cinB = Cin - cinA;
    u32x4 pk;
    #pragma unroll
    for (int j = 0; j < 4; ++j) {
        const int ca = c0 + 2 * j, cb = ca + 1;
        const float va = (ca < cinA) ? xA[((size_t)b * cinA + ca) * HW + s]
                                     : xB[((size_t)b * cinB + (ca - cinA)) * HW + s];
        const float vb = (cb < cinA) ? xA[((size_t)b * cinA + cb) * HW + s]
                                     : xB[((size_t)b * cinB + (cb - cinA)) * HW + s];
        pk[j] = (unsigned)f2bf(va) | ((unsigned)f2bf(vb) << 16);
    }
    *(u32x4*)(out + p * Cin + c0) = pk;
}

// ---------------- bilinear x2 upsample -> NHWC bf16 channel slice ----------------
__global__ __launch_bounds__(256) void up2_nhwc_kernel(
    const float* __restrict__ in, unsigned short* __restrict__ outH,
    int Bn, int C, int H, int W, int hCtot, int hOff, float scale)
{
    const int oH = 2 * H, oW = 2 * W;
    const size_t total = (size_t)Bn * oH * oW * C;
    size_t i = (size_t)blockIdx.x * blockDim.x + threadIdx.x;
    if (i >= total) return;
    const int c = (int)(i % C);
    size_t p = i / C;
    const int ox = (int)(p % oW);
    size_t t = p / oW;
    const int oy = (int)(t % oH);
    const int b = (int)(t / oH);

    float sy = (oy + 0.5f) * 0.5f - 0.5f; sy = sy > 0.f ? sy : 0.f;
    float sx = (ox + 0.5f) * 0.5f - 0.5f; sx = sx > 0.f ? sx : 0.f;
    const int y0 = (int)sy; const int y1 = y0 + 1 < H ? y0 + 1 : H - 1; const float wy = sy - (float)y0;
    const int x0 = (int)sx; const int x1 = x0 + 1 < W ? x0 + 1 : W - 1; const float wx = sx - (float)x0;

    const float* src = in + ((size_t)b * C + c) * H * W;
    const float v0 = src[y0 * W + x0] * (1.f - wx) + src[y0 * W + x1] * wx;
    const float v1 = src[y1 * W + x0] * (1.f - wx) + src[y1 * W + x1] * wx;
    outH[p * hCtot + hOff + c] = f2bf((v0 * (1.f - wy) + v1 * wy) * scale);
}

// ---------------- implicit-GEMM 3x3 conv, NHWC bf16 in, WMMA ----------------
// Block = 256 thr = 8 waves; tile = 64 cout x 32 pixels (one row segment).
// Stage 3 x 34 x CIN halo tile in LDS once (TDM if available), then 9*CIN/32 WMMAs/wave.
template<int CIN>
__global__ __launch_bounds__(256) void conv3x3_nhwc_wmma(
    const unsigned short* __restrict__ xh,      // [B,H,W,CIN] bf16
    const unsigned short* __restrict__ Wp,      // [Coutpad][9*CIN] bf16
    const float* __restrict__ bias,
    float* __restrict__ outN, int nCtot, int nOff,        // NCHW fp32 (optional)
    unsigned short* __restrict__ outH, int hCtot, int hOff, // NHWC bf16 (optional)
    int Cout, int Bn, int H, int W, int act)
{
    __shared__ unsigned short Xs[3 * 34 * CIN];

    constexpr int K   = 9 * CIN;
    constexpr int VEC = CIN / 8;
    const int tid  = threadIdx.x;
    const int lane = tid & 31, wave = tid >> 5;
    const int wm = wave >> 1, wn = wave & 1;

    const int tpr = W >> 5;                      // tiles per row (W multiple of 32)
    const int bi  = blockIdx.x;
    const int x0  = (bi % tpr) * 32;
    const int row = bi / tpr;
    const int y   = row % H;
    const int b   = row / H;
    const int m0  = blockIdx.y * 64;

#ifdef HAVE_TDM
    {
        // zero out-of-image halo strips
        const u32x4 zv = {0u, 0u, 0u, 0u};
        if (y == 0)     for (int i = tid; i < 34 * VEC; i += 256) *(u32x4*)&Xs[i * 8] = zv;
        if (y == H - 1) for (int i = tid; i < 34 * VEC; i += 256) *(u32x4*)&Xs[2 * 34 * CIN + i * 8] = zv;
        if (x0 == 0)      for (int r = tid; r < 3 * VEC; r += 256) { int d = r / VEC, cv = r - d * VEC; *(u32x4*)&Xs[(d * 34) * CIN + cv * 8] = zv; }
        if (x0 + 32 == W) for (int r = tid; r < 3 * VEC; r += 256) { int d = r / VEC, cv = r - d * VEC; *(u32x4*)&Xs[(d * 34 + 33) * CIN + cv * 8] = zv; }

        if (wave == 0) {
            const int ys = (y - 1 < 0) ? 0 : y - 1;
            const int ye = (y + 1 >= H) ? H - 1 : y + 1;
            const int xs = (x0 - 1 < 0) ? 0 : x0 - 1;
            const int xe = (x0 + 32 >= W) ? W - 1 : x0 + 32;
            const unsigned nrows = (unsigned)(ye - ys + 1);
            const unsigned tile0 = (unsigned)((xe - xs + 1) * CIN);     // elements (2B each)
            const unsigned long long ga =
                (unsigned long long)(size_t)(xh + ((size_t)((b * H + ys) * W + xs)) * CIN);
            const unsigned lds_addr = (unsigned)(size_t)(&Xs[0]) +
                (unsigned)((((ys - (y - 1)) * 34) + (xs - (x0 - 1))) * CIN) * 2u;
            const unsigned long long str0 = (unsigned long long)W * CIN; // dim0 stride, elements

            u32x4 g0;
            g0[0] = 1u;                                  // count=1 descriptor
            g0[1] = lds_addr;                            // lds_addr [63:32]
            g0[2] = (unsigned)(ga & 0xFFFFFFFFull);      // global_addr lo
            g0[3] = (unsigned)((ga >> 32) & 0x1FFFFFFull) | 0x80000000u;  // ga hi + type=2
            i32x8 g1;
            g1[0] = (int)(1u << 16);                     // data_size=1 (2 bytes/elem)
            g1[1] = (int)((tile0 & 0xFFFFu) << 16);      // tensor_dim0 lo16 @ [63:48]
            g1[2] = (int)(((tile0 >> 16) & 0xFFFFu) | ((nrows & 0xFFFFu) << 16)); // dim0 hi, dim1 lo
            g1[3] = (int)((tile0 & 0xFFFFu) << 16);      // dim1 hi16 (=0) | tile_dim0 @ [127:112]
            g1[4] = (int)(nrows & 0xFFFFu);              // tile_dim1, tile_dim2=0
            g1[5] = (int)(str0 & 0xFFFFFFFFull);         // tensor_dim0_stride lo32
            g1[6] = (int)((str0 >> 32) & 0xFFFFull);     // stride hi16, dim1_stride lo=0
            g1[7] = 0;
            const i32x4 gz4 = {0, 0, 0, 0};
            const i32x8 gz8 = {0, 0, 0, 0, 0, 0, 0, 0};
            __builtin_amdgcn_tensor_load_to_lds(g0, g1, gz4, gz4, gz8, 0);
            __builtin_amdgcn_s_wait_tensorcnt(0);
        }
    }
#else
    {
        const int totalVec = 102 * VEC;               // 3*34 pixels
        for (int idx = tid; idx < totalVec; idx += 256) {
            const int pixi = idx / VEC;
            const int cv   = idx - pixi * VEC;
            const int dyr  = pixi / 34;
            const int cx   = pixi - dyr * 34;
            const int yy = y - 1 + dyr, xx = x0 - 1 + cx;
            u32x4 v = {0u, 0u, 0u, 0u};
            if ((unsigned)yy < (unsigned)H && (unsigned)xx < (unsigned)W)
                v = *(const u32x4*)(xh + (((size_t)(b * H + yy) * W + xx)) * CIN + cv * 8);
            *(u32x4*)(&Xs[(dyr * 34 + cx) * CIN + cv * 8]) = v;
        }
    }
#endif
    __syncthreads();

    // A addressing (contiguous k = 16*(lane/16) + s, same order as B), B from LDS (per-pixel contiguous K)
    const unsigned short* wr = Wp + (size_t)(m0 + wm * 16 + (lane & 15)) * K;
    const int asel = (lane >> 4) * 16;
    const int bloc = wn * 16 + (lane & 15);      // local pixel 0..31
    const int bsel = (lane >> 4) * 16;

    v8f acc = {};
    #pragma unroll
    for (int dy = 0; dy < 3; ++dy) {
        #pragma unroll
        for (int dx = 0; dx < 3; ++dx) {
            #pragma unroll
            for (int c0 = 0; c0 < CIN; c0 += 32) {
                const int kbase = (dy * 3 + dx) * CIN + c0;
                __builtin_prefetch(wr + kbase + 64, 0, 1);
                AFrag af;
                af.u.lo = *(const u32x4*)(wr + kbase + asel);
                af.u.hi = *(const u32x4*)(wr + kbase + 8 + asel);
                const v16bf bfrag = *(const v16bf*)(&Xs[(dy * 34 + bloc + dx) * CIN + c0 + bsel]);
                acc = __builtin_amdgcn_wmma_f32_16x16x32_bf16(
                    false, af.v, false, bfrag, (short)0, acc, false, false);
            }
        }
    }

    // epilogue
    const int n = lane & 15;
    const int mhalf = (lane >> 4) * 8;
    const int coutBase = m0 + wm * 16 + mhalf;
    const int px = x0 + wn * 16 + n;
    float vals[8];
    #pragma unroll
    for (int r = 0; r < 8; ++r) {
        const int cout = coutBase + r;
        float v = (cout < Cout) ? (acc[r] + bias[cout]) : 0.f;
        if (act) v = (v >= 0.f) ? v : 0.1f * v;
        vals[r] = v;
    }
    const size_t pixIdx = (size_t)(b * H + y) * W + px;
    if (outN) {
        #pragma unroll
        for (int r = 0; r < 8; ++r)
            if (coutBase + r < Cout)
                outN[((size_t)b * nCtot + nOff + coutBase + r) * H * W + y * W + px] = vals[r];
    }
    if (outH && coutBase + 7 < Cout) {
        u32x4 pk;
        #pragma unroll
        for (int j = 0; j < 4; ++j)
            pk[j] = (unsigned)f2bf(vals[2 * j]) | ((unsigned)f2bf(vals[2 * j + 1]) << 16);
        *(u32x4*)(outH + pixIdx * hCtot + hOff + coutBase) = pk;
    }
}

// ---------------- fused DCNv2: stage all K=576 sampled values, then 18 WMMAs ----------------
__global__ __launch_bounds__(256) void dcn_wmma_kernel(
    const float* __restrict__ x, const float* __restrict__ om,
    const unsigned short* __restrict__ Wp, const float* __restrict__ bias,
    float* __restrict__ outN,
    unsigned short* __restrict__ outH, int hCtot, int hOff,
    int Bn, int H, int W, int act)
{
    constexpr int BSTR = 592;                    // 576 + pad, 1184B row (32B aligned)
    __shared__ unsigned short Bt[32 * BSTR];

    const int HW  = H * W;
    const int BHW = Bn * HW;
    const int K   = 576;
    const int p0  = blockIdx.x * 32;
    const int tid = threadIdx.x;
    const int lane = tid & 31, wave = tid >> 5;
    const int wm = wave >> 1, wn = wave & 1;

    const int pix  = tid >> 3;
    const int kgrp = tid & 7;
    const int fp = p0 + pix;
    const bool pvalid = fp < BHW;
    int fb = 0, fy = 0, fx = 0, fs = 0;
    if (pvalid) { fb = fp / HW; fs = fp - fb * HW; fy = fs / W; fx = fs - fy * W; }

    for (int chunk = 0; chunk < 18; ++chunk) {
        const int K0 = chunk * 32;
        #pragma unroll
        for (int j = 0; j < 4; ++j) {
            const int kloc = K0 + kgrp * 4 + j;
            float val = 0.f;
            if (pvalid) {
                const int cin = kloc / 9;
                const int tap = kloc - cin * 9;
                const int g  = cin >> 3;
                const int ky = tap / 3, kx = tap - ky * 3;
                const size_t obase = (size_t)fb * 216 * HW + fs;
                const float dyv = om[obase + (size_t)(g * 9 + tap) * HW];
                const float dxv = om[obase + (size_t)(72 + g * 9 + tap) * HW];
                const float mlg = om[obase + (size_t)(144 + g * 9 + tap) * HW];
                const float mask = 1.f / (1.f + __expf(-mlg));
                const float py = (float)fy - 1.f + (float)ky + dyv;
                const float px = (float)fx - 1.f + (float)kx + dxv;
                const float y0f = floorf(py), x0f = floorf(px);
                const float wy = py - y0f, wx = px - x0f;
                const int iy0 = (int)y0f, ix0 = (int)x0f;
                const int iy1 = iy0 + 1,  ix1 = ix0 + 1;
                const float* xc = x + ((size_t)fb * 64 + cin) * HW;
                const bool y0v = (unsigned)iy0 < (unsigned)H, y1v = (unsigned)iy1 < (unsigned)H;
                const bool x0v = (unsigned)ix0 < (unsigned)W, x1v = (unsigned)ix1 < (unsigned)W;
                float a = 0.f;
                if (y0v && x0v) a += xc[iy0 * W + ix0] * (1.f - wy) * (1.f - wx);
                if (y0v && x1v) a += xc[iy0 * W + ix1] * (1.f - wy) * wx;
                if (y1v && x0v) a += xc[iy1 * W + ix0] * wy * (1.f - wx);
                if (y1v && x1v) a += xc[iy1 * W + ix1] * wy * wx;
                val = a * mask;
            }
            Bt[pix * BSTR + kloc] = f2bf(val);
        }
    }
    __syncthreads();

    const unsigned short* wr = Wp + (size_t)(wm * 16 + (lane & 15)) * K;
    const int asel = (lane >> 4) * 16;
    const int bcol = wn * 16 + (lane & 15);
    const int bsel = (lane >> 4) * 16;

    v8f acc = {};
    #pragma unroll
    for (int chunk = 0; chunk < 18; ++chunk) {
        const int K0 = chunk * 32;
        __builtin_prefetch(wr + K0 + 64, 0, 1);
        AFrag af;
        af.u.lo = *(const u32x4*)(wr + K0 + asel);
        af.u.hi = *(const u32x4*)(wr + K0 + 8 + asel);
        const v16bf bfrag = *(const v16bf*)(&Bt[bcol * BSTR + K0 + bsel]);
        acc = __builtin_amdgcn_wmma_f32_16x16x32_bf16(
            false, af.v, false, bfrag, (short)0, acc, false, false);
    }

    const int n = lane & 15;
    const int mhalf = (lane >> 4) * 8;
    const int coutBase = wm * 16 + mhalf;
    const int pp = p0 + wn * 16 + n;
    if (pp < BHW) {
        const int ob = pp / HW;
        const int os = pp - ob * HW;
        float vals[8];
        #pragma unroll
        for (int r = 0; r < 8; ++r) {
            float v = acc[r] + bias[coutBase + r];
            if (act) v = (v >= 0.f) ? v : 0.1f * v;
            vals[r] = v;
        }
        if (outN) {
            #pragma unroll
            for (int r = 0; r < 8; ++r)
                outN[((size_t)ob * 64 + coutBase + r) * HW + os] = vals[r];
        }
        if (outH) {
            u32x4 pk;
            #pragma unroll
            for (int j = 0; j < 4; ++j)
                pk[j] = (unsigned)f2bf(vals[2 * j]) | ((unsigned)f2bf(vals[2 * j + 1]) << 16);
            *(u32x4*)(outH + (size_t)pp * hCtot + hOff + coutBase) = pk;
        }
    }
}

// ---------------- host orchestration ----------------
extern "C" void kernel_launch(void* const* d_in, const int* in_sizes, int n_in,
                              void* d_out, int out_size, void* d_ws, size_t ws_size,
                              hipStream_t stream) {
    (void)in_sizes; (void)n_in; (void)out_size; (void)ws_size;
    const int Bn = 4;
    const int H1 = 128, W1 = 128, H2 = 64, W2 = 64, H3 = 32, W3 = 32;
    const size_t HW1 = (size_t)H1 * W1, HW2 = (size_t)H2 * W2, HW3 = (size_t)H3 * W3;
    const size_t F2 = Bn * 64 * HW2, F3 = Bn * 64 * HW3;
    const size_t OM1 = Bn * 216 * HW1, OM2 = Bn * 216 * HW2, OM3 = Bn * 216 * HW3;

    // float region
    float* wsf = (float*)d_ws;
    size_t fo = 0;
    auto fal = [&](size_t nelem) { float* p = wsf + fo; fo += (nelem + 63) & ~(size_t)63; return p; };
    float* l3o_n = fal(F3); float* l3f_n = fal(F3); float* om3 = fal(OM3);
    float* l2o_n = fal(F2); float* l2f_n = fal(F2); float* om2 = fal(OM2);
    float* om1   = fal(OM1);

    // bf16 (ushort) region
    unsigned short* wsh = (unsigned short*)(wsf + fo);
    size_t ho = 0;
    auto hal = [&](size_t nelem) { unsigned short* p = wsh + ho; ho += (nelem + 127) & ~(size_t)127; return p; };
    unsigned short* c3cat = hal(Bn * HW3 * 128);
    unsigned short* t3a_h = hal(Bn * HW3 * 64);
    unsigned short* l3o_h = hal(Bn * HW3 * 64);
    unsigned short* cat2a = hal(Bn * HW2 * 128);
    unsigned short* cat2b = hal(Bn * HW2 * 128);
    unsigned short* t2b_h = hal(Bn * HW2 * 64);
    unsigned short* l2o_h = hal(Bn * HW2 * 64);
    unsigned short* catf2 = hal(Bn * HW2 * 128);
    unsigned short* cat1a = hal(Bn * HW1 * 128);
    unsigned short* cat1b = hal(Bn * HW1 * 128);
    unsigned short* t1b_h = hal(Bn * HW1 * 64);
    unsigned short* l1o_h = hal(Bn * HW1 * 64);
    unsigned short* catf1 = hal(Bn * HW1 * 128);

    // weight region
    unsigned short* wb = wsh + ho;
    size_t wo = 0;
    auto packw = [&](int widx, int Cout, int Cin, int cin_major) -> const unsigned short* {
        const int Coutpad = ((Cout + 63) / 64) * 64;
        const size_t nelem = (size_t)Coutpad * 9 * Cin;
        unsigned short* dst = wb + wo;
        wo += (nelem + 127) & ~(size_t)127;
        pack_w_kernel<<<dim3((unsigned)((nelem + 255) / 256)), 256, 0, stream>>>(
            (const float*)d_in[widx], dst, Cout, Cin, Coutpad, cin_major);
        return dst;
    };
    auto cvt = [&](const float* xA, const float* xB, int cinA, int Cin,
                   unsigned short* out, int H, int W) {
        const size_t total = (size_t)Bn * H * W * (Cin / 8);
        nchw_cat_to_nhwc_kernel<<<dim3((unsigned)((total + 255) / 256)), 256, 0, stream>>>(
            xA, xB, cinA, Cin, out, Bn, H, W);
    };
    auto up2 = [&](const float* in, unsigned short* outH, int H, int W,
                   int hCtot, int hOff, float scale) {
        const size_t total = (size_t)Bn * 4 * H * W * 64;
        up2_nhwc_kernel<<<dim3((unsigned)((total + 255) / 256)), 256, 0, stream>>>(
            in, outH, Bn, 64, H, W, hCtot, hOff, scale);
    };
    auto conv = [&](const unsigned short* xh, int Cin, const unsigned short* Wp, int bidx,
                    float* outN, int nCtot, int nOff,
                    unsigned short* outH, int hCtot, int hOff,
                    int Cout, int H, int W, int act) {
        dim3 g((unsigned)(Bn * H * (W / 32)), (unsigned)((Cout + 63) / 64));
        const float* bias = (const float*)d_in[bidx];
        if (Cin == 128)
            conv3x3_nhwc_wmma<128><<<g, 256, 0, stream>>>(xh, Wp, bias, outN, nCtot, nOff,
                                                          outH, hCtot, hOff, Cout, Bn, H, W, act);
        else
            conv3x3_nhwc_wmma<64><<<g, 256, 0, stream>>>(xh, Wp, bias, outN, nCtot, nOff,
                                                         outH, hCtot, hOff, Cout, Bn, H, W, act);
    };
    auto dcn = [&](const float* x, const float* omb, const unsigned short* Wp, int bidx,
                   float* outN, unsigned short* outH, int hCtot, int hOff,
                   int H, int W, int act) {
        const int BHW = Bn * H * W;
        dcn_wmma_kernel<<<dim3((unsigned)((BHW + 31) / 32)), 256, 0, stream>>>(
            x, omb, Wp, (const float*)d_in[bidx], outN, outH, hCtot, hOff, Bn, H, W, act);
    };

    const float* f1[3] = { (const float*)d_in[0], (const float*)d_in[1], (const float*)d_in[2] };
    const float* f2p[3] = { (const float*)d_in[3], (const float*)d_in[4], (const float*)d_in[5] };

    for (int s = 0; s < 2; ++s) {
        // jax pytree (sorted-key) flat leaf indices; params begin at d_in[6]
        const int L1dcn = 6 + s * 4,  L1fea = 14 + s * 2, L1oc1 = 18 + s * 2,
                  L1oc2 = 22 + s * 2, L1oc3 = 26 + s * 2,
                  L2dcn = 30 + s * 4, L2fea = 38 + s * 2, L2oc1 = 42 + s * 2,
                  L2oc2 = 46 + s * 2, L2oc3 = 50 + s * 2,
                  L3dcn = 54 + s * 4, L3oc1 = 62 + s * 2, L3oc2 = 66 + s * 2;
        // dcnpack leaves sorted: b, om_b, om_w, w
        const unsigned short* w_oc1_3 = packw(L3oc1 + 1, 64, 128, 0);
        const unsigned short* w_oc2_3 = packw(L3oc2 + 1, 64, 64, 0);
        const unsigned short* w_om3   = packw(L3dcn + 2, 216, 64, 0);
        const unsigned short* w_dcn3  = packw(L3dcn + 3, 64, 64, 1);
        const unsigned short* w_oc1_2 = packw(L2oc1 + 1, 64, 128, 0);
        const unsigned short* w_oc2_2 = packw(L2oc2 + 1, 64, 128, 0);
        const unsigned short* w_oc3_2 = packw(L2oc3 + 1, 64, 64, 0);
        const unsigned short* w_om2   = packw(L2dcn + 2, 216, 64, 0);
        const unsigned short* w_dcn2  = packw(L2dcn + 3, 64, 64, 1);
        const unsigned short* w_fea2  = packw(L2fea + 1, 64, 128, 0);
        const unsigned short* w_oc1_1 = packw(L1oc1 + 1, 64, 128, 0);
        const unsigned short* w_oc2_1 = packw(L1oc2 + 1, 64, 128, 0);
        const unsigned short* w_oc3_1 = packw(L1oc3 + 1, 64, 64, 0);
        const unsigned short* w_om1   = packw(L1dcn + 2, 216, 64, 0);
        const unsigned short* w_dcn1  = packw(L1dcn + 3, 64, 64, 1);
        const unsigned short* w_fea1  = packw(L1fea + 1, 64, 128, 0);

        const float* const* fa = (s == 0) ? f1 : f2p;
        const float* const* fb = (s == 0) ? f2p : f1;

        // ---- L3 ----
        cvt(fa[2], fb[2], 64, 128, c3cat, H3, W3);
        conv(c3cat, 128, w_oc1_3, L3oc1, nullptr, 0, 0, t3a_h, 64, 0, 64, H3, W3, 1);
        conv(t3a_h, 64, w_oc2_3, L3oc2, l3o_n, 64, 0, l3o_h, 64, 0, 64, H3, W3, 1);
        conv(l3o_h, 64, w_om3, L3dcn + 1, om3, 216, 0, nullptr, 0, 0, 216, H3, W3, 0);
        dcn(fa[2], om3, w_dcn3, L3dcn, l3f_n, nullptr, 0, 0, H3, W3, 1);
        // ---- L2 ----
        cvt(fa[1], fb[1], 64, 128, cat2a, H2, W2);
        conv(cat2a, 128, w_oc1_2, L2oc1, nullptr, 0, 0, cat2b, 128, 0, 64, H2, W2, 1);
        up2(l3o_n, cat2b, H3, W3, 128, 64, 2.0f);
        conv(cat2b, 128, w_oc2_2, L2oc2, nullptr, 0, 0, t2b_h, 64, 0, 64, H2, W2, 1);
        conv(t2b_h, 64, w_oc3_2, L2oc3, l2o_n, 64, 0, l2o_h, 64, 0, 64, H2, W2, 1);
        conv(l2o_h, 64, w_om2, L2dcn + 1, om2, 216, 0, nullptr, 0, 0, 216, H2, W2, 0);
        dcn(fa[1], om2, w_dcn2, L2dcn, nullptr, catf2, 128, 0, H2, W2, 0);
        up2(l3f_n, catf2, H3, W3, 128, 64, 1.0f);
        conv(catf2, 128, w_fea2, L2fea, l2f_n, 64, 0, nullptr, 0, 0, 64, H2, W2, 1);
        // ---- L1 ----
        cvt(fa[0], fb[0], 64, 128, cat1a, H1, W1);
        conv(cat1a, 128, w_oc1_1, L1oc1, nullptr, 0, 0, cat1b, 128, 0, 64, H1, W1, 1);
        up2(l2o_n, cat1b, H2, W2, 128, 64, 2.0f);
        conv(cat1b, 128, w_oc2_1, L1oc2, nullptr, 0, 0, t1b_h, 64, 0, 64, H1, W1, 1);
        conv(t1b_h, 64, w_oc3_1, L1oc3, nullptr, 0, 0, l1o_h, 64, 0, 64, H1, W1, 1);
        conv(l1o_h, 64, w_om1, L1dcn + 1, om1, 216, 0, nullptr, 0, 0, 216, H1, W1, 0);
        dcn(fa[0], om1, w_dcn1, L1dcn, nullptr, catf1, 128, 0, H1, W1, 0);
        up2(l2f_n, catf1, H2, W2, 128, 64, 1.0f);
        // final conv writes directly into d_out channel slice [s*64, s*64+64)
        conv(catf1, 128, w_fea1, L1fea, (float*)d_out, 128, s * 64, nullptr, 0, 0, 64, H1, W1, 0);
    }
}
